// QuantileCalibrator_23141283791296
// MI455X (gfx1250) — compile-verified, matrix operand order checked by reading, not run on hardware
//
#include <hip/hip_runtime.h>
#include <hip/hip_bf16.h>

// ---------------------------------------------------------------------------
// QuantileCalibrator fused kernel for MI455X (gfx1250, wave32, WMMA).
//
// Roofline: ~238 GFLOP of GEMM vs ~640MB of HBM traffic (27us @ 23.3TB/s).
// Strategy: f16 WMMA (v_wmma_f32_16x16x32_f16, f32 accumulate) for all GEMMs;
// weights pre-packed once into per-lane B-fragment layout (contiguous 32B per
// lane -> two global_load_b128); activations live in LDS with strides chosen
// so every A-fragment is two 16B ds loads; one fused kernel per 48-row tile
// (== one batch element, so h_emb index == local row).
// ---------------------------------------------------------------------------

typedef __attribute__((ext_vector_type(16))) _Float16 v16h;
typedef __attribute__((ext_vector_type(8)))  float    v8f;

#define B_SZ   4096
#define L_SZ   256
#define D_SZ   32
#define H_SZ   48
#define M_SZ   64
#define HID    512
#define INDIM  96          // M + D
#define ROWS   48          // rows per workgroup (== H)
#define LDZ    520         // LDS stride (halfs), multiple of 8 -> 16B-aligned frags
#define LDF    104         // LDS stride (halfs) for feats, multiple of 8
#define LDA    68          // LDS stride (floats) for anchor
#define NW     8           // waves per block
#define BLK    256

__device__ __forceinline__ float gelu_exact(float x) {
  return 0.5f * x * (1.0f + erff(x * 0.70710678118654752f));
}
__device__ __forceinline__ float softplus_f(float x) {
  return (x > 20.0f) ? x : log1pf(expf(x));
}

// WMMA: D = A(16x32 f16) * B(32x16 f16) + C(16x16 f32)
__device__ __forceinline__ v8f wmma16(v16h a, v16h b, v8f c) {
  return __builtin_amdgcn_wmma_f32_16x16x32_f16(false, a, false, b,
                                                (short)0, c, false, false);
}

// A-fragment (16x32, f16) from LDS, row-major with stride ld (halfs).
// Lane layout: lanes 0-15 rows 0-15 (k 0..7 & 16..23), lanes 16-31 same rows
// with k offset +8. Each group of 8 halfs is contiguous and 16B aligned
// (ld % 8 == 0, kbase % 8 == 0) -> two ds b128 loads per fragment.
__device__ __forceinline__ v16h load_a_lds(const _Float16* __restrict__ base,
                                           int lane, int kbase, int ld) {
  union { v16h v; uint4 q[2]; } f;
  const int r = lane & 15, hlf = lane >> 4;
  const _Float16* p = base + r * ld + kbase + 8 * hlf;
  f.q[0] = *(const uint4*)(p);        // k .. k+7
  f.q[1] = *(const uint4*)(p + 16);   // k+16 .. k+23
  return f.v;
}

// B-fragment (32x16, f16) from pre-packed weights: 512 halfs per tile,
// lane-major (32B contiguous per lane -> two global b128 loads).
__device__ __forceinline__ v16h load_b_pk(const _Float16* __restrict__ pk,
                                          int tile, int lane) {
  return *(const v16h*)(pk + (size_t)tile * 512 + lane * 16);
}

// ---------------------------------------------------------------------------
// Kernel 0: x_pool = mean_L(x)   [B,L,D] -> [B,D]
// ---------------------------------------------------------------------------
__global__ __launch_bounds__(256) void pool_x_kernel(const float* __restrict__ x,
                                                     float* __restrict__ xp) {
  __shared__ float red[256];
  const int b = blockIdx.x, t = threadIdx.x;
  const int d = t & 31, seg = t >> 5;
  const float* px = x + (size_t)b * (L_SZ * D_SZ);
  float s = 0.f;
  for (int l = seg; l < L_SZ; l += 8) s += px[l * D_SZ + d];
  red[t] = s;
  __syncthreads();
  if (t < 32) {
    float tot = 0.f;
#pragma unroll
    for (int k = 0; k < 8; ++k) tot += red[k * 32 + t];
    xp[b * D_SZ + t] = tot * (1.0f / 256.0f);
  }
}

// ---------------------------------------------------------------------------
// Kernel 1: pack a row-major f32 weight [K,N] into f16 B-fragment tiles.
// Tile index = nt*(K/32) + kt ; within tile: lane*16 + i*2 + p maps to
// element (k = kt*32 + 2i + (i>=4?8:0) + 8*(lane>>4) + p, n = nt*16 + (lane&15)).
// ---------------------------------------------------------------------------
__global__ __launch_bounds__(256) void pack_w_kernel(const float* __restrict__ src,
                                                     _Float16* __restrict__ dst,
                                                     int K, int N) {
  const int total = K * N;
  const int KT = K >> 5;
  for (int e = blockIdx.x * blockDim.x + threadIdx.x; e < total;
       e += gridDim.x * blockDim.x) {
    int tile = e >> 9, rem = e & 511;
    int l = rem >> 4, idx = rem & 15, i = idx >> 1, p = idx & 1;
    int nt = tile / KT, kt = tile - nt * KT;
    int k = kt * 32 + 2 * i + ((i >= 4) ? 8 : 0) + ((l >> 4) << 3) + p;
    int n = nt * 16 + (l & 15);
    dst[e] = (_Float16)src[k * N + n];
  }
}

// ---------------------------------------------------------------------------
// Kernel 2: fused trunk + heads + monotone quantile construction.
// One workgroup = 48 rows = one batch element. 8 waves of 32.
// ---------------------------------------------------------------------------
__global__ __launch_bounds__(BLK) void qc_fused_kernel(
    const float* __restrict__ y_hat, const float* __restrict__ x_pool,
    const _Float16* __restrict__ w1p, const _Float16* __restrict__ w2p,
    const _Float16* __restrict__ wap, const _Float16* __restrict__ wup,
    const _Float16* __restrict__ wdp,
    const float* __restrict__ b1, const float* __restrict__ b2,
    const float* __restrict__ h_emb, const float* __restrict__ ba,
    const float* __restrict__ bu, const float* __restrict__ bd,
    float* __restrict__ out) {
  __shared__ __align__(16) _Float16 Xs[ROWS * LDZ];   // z1, then z (in place)
  __shared__ __align__(16) float    FAs[ROWS * LDA];  // feats (f16), then anchor (f32)
  _Float16* feats = (_Float16*)FAs;

  const int tid  = threadIdx.x;
  const int lane = tid & 31;
  const int wave = tid >> 5;
  const int wg   = blockIdx.x;                 // == batch index b
  const long grow0 = (long)wg * ROWS;          // global row base (b*H)

  // ---- stage feats = [y_hat | x_pool] as f16 ------------------------------
  for (int idx = tid; idx < ROWS * INDIM; idx += BLK) {
    int r = idx / INDIM, c = idx - r * INDIM;
    float v = (c < M_SZ) ? y_hat[(grow0 + r) * M_SZ + c]
                         : x_pool[wg * D_SZ + (c - M_SZ)];
    feats[r * LDF + c] = (_Float16)v;
  }
  __syncthreads();

  // ---- GEMM1: z1 = gelu(feats @ W1 + b1), K=96 ----------------------------
  {
    v8f acc[12] = {};
    for (int kt = 0; kt < 3; ++kt) {
      v16h a0 = load_a_lds(feats + 0 * 16 * LDF, lane, kt * 32, LDF);
      v16h a1 = load_a_lds(feats + 1 * 16 * LDF, lane, kt * 32, LDF);
      v16h a2 = load_a_lds(feats + 2 * 16 * LDF, lane, kt * 32, LDF);
#pragma unroll
      for (int c = 0; c < 4; ++c) {
        v16h bfr = load_b_pk(w1p, (wave + 8 * c) * 3 + kt, lane);
        acc[c * 3 + 0] = wmma16(a0, bfr, acc[c * 3 + 0]);
        acc[c * 3 + 1] = wmma16(a1, bfr, acc[c * 3 + 1]);
        acc[c * 3 + 2] = wmma16(a2, bfr, acc[c * 3 + 2]);
      }
    }
#pragma unroll
    for (int c = 0; c < 4; ++c) {
      int col = (wave + 8 * c) * 16 + (lane & 15);
      float bias = b1[col];
#pragma unroll
      for (int rt = 0; rt < 3; ++rt)
#pragma unroll
        for (int v = 0; v < 8; ++v) {
          int row = rt * 16 + v + 8 * (lane >> 4);
          Xs[row * LDZ + col] = (_Float16)gelu_exact(acc[c * 3 + rt][v] + bias);
        }
    }
  }
  __syncthreads();

  // ---- GEMM2: z = gelu(z1 @ W2 + b2) + h_emb, K=512 (in-place via regs) ---
  {
    v8f acc[12] = {};
    for (int kt = 0; kt < 16; ++kt) {
      v16h a0 = load_a_lds(Xs + 0 * 16 * LDZ, lane, kt * 32, LDZ);
      v16h a1 = load_a_lds(Xs + 1 * 16 * LDZ, lane, kt * 32, LDZ);
      v16h a2 = load_a_lds(Xs + 2 * 16 * LDZ, lane, kt * 32, LDZ);
#pragma unroll
      for (int c = 0; c < 4; ++c) {
        v16h bfr = load_b_pk(w2p, (wave + 8 * c) * 16 + kt, lane);
        acc[c * 3 + 0] = wmma16(a0, bfr, acc[c * 3 + 0]);
        acc[c * 3 + 1] = wmma16(a1, bfr, acc[c * 3 + 1]);
        acc[c * 3 + 2] = wmma16(a2, bfr, acc[c * 3 + 2]);
      }
    }
    __syncthreads();   // all z1 reads complete before overwriting Xs with z
#pragma unroll
    for (int c = 0; c < 4; ++c) {
      int col = (wave + 8 * c) * 16 + (lane & 15);
      float bias = b2[col];
#pragma unroll
      for (int rt = 0; rt < 3; ++rt)
#pragma unroll
        for (int v = 0; v < 8; ++v) {
          int row = rt * 16 + v + 8 * (lane >> 4);   // row == h (ROWS == H)
          float t = gelu_exact(acc[c * 3 + rt][v] + bias) + h_emb[row * HID + col];
          Xs[row * LDZ + col] = (_Float16)t;
        }
    }
  }
  __syncthreads();

  // ---- Head A: anchor = y_hat + tanh(z @ Wa + ba) * 0.1 -------------------
  for (int t = wave; t < 12; t += NW) {           // 4 col-tiles x 3 row-tiles
    int ct = t & 3, rt = t >> 2;
    v8f acc = {};
    for (int kt = 0; kt < 16; ++kt) {
      v16h a   = load_a_lds(Xs + rt * 16 * LDZ, lane, kt * 32, LDZ);
      v16h bfr = load_b_pk(wap, ct * 16 + kt, lane);
      acc = wmma16(a, bfr, acc);
    }
    int m = ct * 16 + (lane & 15);
    float bias = ba[m];
#pragma unroll
    for (int v = 0; v < 8; ++v) {
      int row = rt * 16 + v + 8 * (lane >> 4);
      long grow = grow0 + row;
      float anc = y_hat[grow * M_SZ + m] + tanhf(acc[v] + bias) * 0.1f;
      FAs[row * LDA + m] = anc;
      out[(grow * M_SZ + m) * 9 + 4] = anc;       // anchor at j = ANCHOR_IDX
    }
  }
  __syncthreads();

  // ---- Heads U/D: softplus + 4-lane cumsum -> quantiles -------------------
  for (int g = wave; g < 32; g += NW) {           // 16 up tiles + 16 down tiles
    const bool up = (g < 16);
    const int nt = g & 15;
    const _Float16* wpk   = up ? wup : wdp;
    const float*    biasp = up ? bu  : bd;
    v8f acc[3] = {};
    for (int kt = 0; kt < 16; ++kt) {
      v16h bfr = load_b_pk(wpk, nt * 16 + kt, lane);
#pragma unroll
      for (int rt = 0; rt < 3; ++rt) {
        v16h a = load_a_lds(Xs + rt * 16 * LDZ, lane, kt * 32, LDZ);
        acc[rt] = wmma16(a, bfr, acc[rt]);
      }
    }
    int col = nt * 16 + (lane & 15);              // 0..255
    int m = col >> 2, j = col & 3;
    float bias = biasp[col];
#pragma unroll
    for (int rt = 0; rt < 3; ++rt)
#pragma unroll
      for (int v = 0; v < 8; ++v) {
        int row = rt * 16 + v + 8 * (lane >> 4);
        float sp = softplus_f(acc[rt][v] + bias);
        // inclusive scan over j (4 lanes): S[j] = sum_{i<=j} sp_i
        float tshf = __shfl_up(sp, 1, 4);
        float cum = sp + ((j >= 1) ? tshf : 0.0f);
        tshf = __shfl_up(cum, 2, 4);
        cum += (j >= 2) ? tshf : 0.0f;
        float anc = FAs[row * LDA + m];
        long grow = grow0 + row;
        float q = up ? (anc + cum) : (anc - cum);
        int jpos = up ? (5 + j) : (3 - j);        // flip for the down branch
        out[(grow * M_SZ + m) * 9 + jpos] = q;
      }
  }
}

// ---------------------------------------------------------------------------
extern "C" void kernel_launch(void* const* d_in, const int* in_sizes, int n_in,
                              void* d_out, int out_size, void* d_ws, size_t ws_size,
                              hipStream_t stream) {
  (void)in_sizes; (void)n_in; (void)out_size; (void)ws_size;
  const float* x     = (const float*)d_in[0];
  const float* y_hat = (const float*)d_in[1];
  const float* W1    = (const float*)d_in[2];
  const float* b1    = (const float*)d_in[3];
  const float* W2    = (const float*)d_in[4];
  const float* b2    = (const float*)d_in[5];
  const float* h_emb = (const float*)d_in[6];
  const float* Wa    = (const float*)d_in[7];
  const float* ba    = (const float*)d_in[8];
  const float* Wu    = (const float*)d_in[9];
  const float* bu    = (const float*)d_in[10];
  const float* Wd    = (const float*)d_in[11];
  const float* bd    = (const float*)d_in[12];
  float* out = (float*)d_out;

  // Workspace layout (all offsets 32B aligned):
  char* ws = (char*)d_ws;
  float*    x_pool = (float*)ws;                        // 4096*32*4  = 524288
  _Float16* w1p    = (_Float16*)(ws + 524288);          //  96*512*2  =  98304
  _Float16* w2p    = (_Float16*)(ws + 622592);          // 512*512*2  = 524288
  _Float16* wap    = (_Float16*)(ws + 1146880);         // 512*64*2   =  65536
  _Float16* wup    = (_Float16*)(ws + 1212416);         // 512*256*2  = 262144
  _Float16* wdp    = (_Float16*)(ws + 1474560);         // 512*256*2  = 262144

  pool_x_kernel<<<B_SZ, 256, 0, stream>>>(x, x_pool);
  pack_w_kernel<<<(INDIM * HID + 255) / 256, 256, 0, stream>>>(W1, w1p, INDIM, HID);
  pack_w_kernel<<<(HID * HID + 255) / 256, 256, 0, stream>>>(W2, w2p, HID, HID);
  pack_w_kernel<<<(HID * M_SZ + 255) / 256, 256, 0, stream>>>(Wa, wap, HID, M_SZ);
  pack_w_kernel<<<(HID * 256 + 255) / 256, 256, 0, stream>>>(Wu, wup, HID, 256);
  pack_w_kernel<<<(HID * 256 + 255) / 256, 256, 0, stream>>>(Wd, wdp, HID, 256);

  qc_fused_kernel<<<B_SZ, BLK, 0, stream>>>(y_hat, x_pool, w1p, w2p, wap, wup, wdp,
                                            b1, b2, h_emb, ba, bu, bd, out);
}
